// MultiHeadAttention_15745350107388
// MI455X (gfx1250) — matrix operand order repaired and checked by reading, not compiled, not run on hardware
//
#include <hip/hip_runtime.h>
#include <hip/hip_bf16.h>

// Problem constants (match reference)
#define B_  2
#define T_  2048
#define D_  1024
#define H_  16
#define HS_ 64

typedef __attribute__((ext_vector_type(16))) _Float16 v16h;
typedef __attribute__((ext_vector_type(8)))  _Float16 v8h;
typedef __attribute__((ext_vector_type(8)))  float    v8f;
typedef __attribute__((ext_vector_type(4)))  unsigned int v4u;
typedef __attribute__((ext_vector_type(8)))  int v8i;
typedef __attribute__((ext_vector_type(4)))  int v4i;

union ABFrag { v16h v; v8h h[2]; };

__device__ __forceinline__ v8f wmma16(v16h a, v16h b, v8f c) {
  // (neg_a, A, neg_b, B, c_mod, C, reuse_a, reuse_b)
  return __builtin_amdgcn_wmma_f32_16x16x32_f16(false, a, false, b, (short)0, c,
                                                false, false);
}

// --------------------------------------------------------------------------
// Tensor Data Mover: 2D tile load (global -> LDS), f16 elements.
// D# per CDNA5 ISA ch.8: group0 = {count/lds_addr/global_addr/type=2},
// group1 = {data_size=1(2B), tensor_dim0/1, tile_dim0/1, dim0_stride}.
// dims/strides in elements, addresses in bytes. This toolchain exposes the
// 6-arg builtin: (g0, g1, g2, g3, g4, cpol); groups 2/3/4 unused for 2D.
// --------------------------------------------------------------------------
__device__ __forceinline__ void tdm_load_2d_f16(unsigned int lds_off,
                                                const _Float16* gptr,
                                                unsigned int dim0,
                                                unsigned int dim1,
                                                unsigned int tile0,
                                                unsigned int tile1,
                                                unsigned int stride0) {
  unsigned long long ga = (unsigned long long)gptr;
  v4u g0;
  g0[0] = 1u;                                   // count=1, user descriptor
  g0[1] = lds_off;                              // LDS byte address
  g0[2] = (unsigned int)ga;                     // global_addr[31:0]
  g0[3] = (unsigned int)((ga >> 32) & 0x1FFFFFFu) | (2u << 30); // addr[56:32]|type=2
  v8i g1;
  g1[0] = (int)(1u << 16);                      // wg_mask=0, data_size=1 (2B)
  g1[1] = (int)((dim0 & 0xFFFFu) << 16);        // tensor_dim0[15:0] @ bits63:48
  g1[2] = (int)((dim0 >> 16) | ((dim1 & 0xFFFFu) << 16)); // dim0[31:16]|dim1[15:0]
  g1[3] = (int)((dim1 >> 16) | (tile0 << 16));  // dim1[31:16] | tile_dim0
  g1[4] = (int)(tile1 & 0xFFFFu);               // tile_dim1 (tile_dim2=0)
  g1[5] = (int)stride0;                         // tensor_dim0_stride[31:0]
  g1[6] = 0;                                    // stride0[47:32], dim1_stride lo
  g1[7] = 0;                                    // dim1_stride hi (unused, 2D)
  v4i z4 = {0, 0, 0, 0};
  v8i z8 = {0, 0, 0, 0, 0, 0, 0, 0};
  __builtin_amdgcn_tensor_load_to_lds(g0, g1, z4, z4, z8, 0);
}

// ---------------------------------------------------------------- cast q/k/v
__global__ void cast_qkv_kernel(const float* __restrict__ q,
                                const float* __restrict__ k,
                                const float* __restrict__ v,
                                _Float16* __restrict__ qo,
                                _Float16* __restrict__ ko,
                                _Float16* __restrict__ vo) {
  const int n = B_ * T_ * D_;
  int i = blockIdx.x * blockDim.x + threadIdx.x;
  if (i < n) {
    qo[i] = (_Float16)q[i];
    ko[i] = (_Float16)k[i];
    vo[i] = (_Float16)v[i];
  }
}

// ------------------------------------------ cast + transpose weight matrices
// Wq/Wk/Wv: [H,D,HS] -> [H,HS,D]   Wo: [D,D] -> WoT[n][k] = Wo[k][n]
__global__ void cast_w_kernel(const float* __restrict__ Wq,
                              const float* __restrict__ Wk,
                              const float* __restrict__ Wv,
                              const float* __restrict__ Wo,
                              _Float16* __restrict__ WqT,
                              _Float16* __restrict__ WkT,
                              _Float16* __restrict__ WvT,
                              _Float16* __restrict__ WoT) {
  const int nW = H_ * D_ * HS_;  // == D_*D_
  int i = blockIdx.x * blockDim.x + threadIdx.x;
  if (i < nW) {
    int h  = i / (D_ * HS_);
    int r  = i % (D_ * HS_);
    int kk = r / HS_;
    int n  = r % HS_;
    int o  = (h * HS_ + n) * D_ + kk;
    WqT[o] = (_Float16)Wq[i];
    WkT[o] = (_Float16)Wk[i];
    WvT[o] = (_Float16)Wv[i];
    int ko2 = i / D_;
    int no2 = i % D_;
    WoT[no2 * D_ + ko2] = (_Float16)Wo[i];
  }
}

// ------------------------------------------------- per-head QKV projections
// One wave computes a 64(t) x 64(hs) tile: 16 accumulators, 16 WMMAs per
// 8 fragment fetches (4x A, 4x B) -> 4x the L2 reuse of a 16x64 tile.
// Q is pre-scaled by HS^-0.5.  V is stored transposed: VpT[b][h][hs][t].
__global__ void __launch_bounds__(256) proj_kernel(
    const _Float16* __restrict__ qh, const _Float16* __restrict__ kh,
    const _Float16* __restrict__ vh,
    const _Float16* __restrict__ WqT, const _Float16* __restrict__ WkT,
    const _Float16* __restrict__ WvT,
    _Float16* __restrict__ Qp, _Float16* __restrict__ Kp,
    _Float16* __restrict__ VpT) {
  const int lane = threadIdx.x & 31;
  const int wave = threadIdx.x >> 5;
  const int gw = blockIdx.x * 8 + wave;

  const int wavesPerMat = B_ * H_ * (T_ / 64);  // 1024
  const int which = gw / wavesPerMat;           // 0=Q 1=K 2=V
  const int rem   = gw % wavesPerMat;
  const int b  = rem / (H_ * (T_ / 64));
  const int h  = (rem / (T_ / 64)) % H_;
  const int mt = rem % (T_ / 64);
  const int tBase = mt * 64;

  const _Float16* X  = (which == 0) ? qh  : (which == 1) ? kh  : vh;
  const _Float16* WT = (which == 0) ? WqT : (which == 1) ? WkT : WvT;

  const int n0  = lane & 15;       // A row / B column within a 16-subtile
  const int hi  = lane >> 4;       // half-wave select
  const int kbA = hi ? 8  : 0;
  const int kbB = hi ? 16 : 0;

  const _Float16* Xbase = X + ((size_t)b * T_ + tBase) * D_;
  const _Float16* Wbase = WT + (size_t)h * HS_ * D_;

  v8f acc[4][4];  // [m-subtile][n-subtile]
  #pragma unroll
  for (int ms = 0; ms < 4; ++ms)
    #pragma unroll
    for (int nt = 0; nt < 4; ++nt)
      #pragma unroll
      for (int r = 0; r < 8; ++r) acc[ms][nt][r] = 0.f;

  for (int kc = 0; kc < D_ / 32; ++kc) {
    const int k0 = kc * 32;
    ABFrag a[4];
    #pragma unroll
    for (int ms = 0; ms < 4; ++ms) {
      const _Float16* Xrow = Xbase + (size_t)(ms * 16 + n0) * D_;
      a[ms].h[0] = *(const v8h*)(Xrow + k0 + kbB);
      a[ms].h[1] = *(const v8h*)(Xrow + k0 + kbB + 8);
    }
    v16h bf[4];
    #pragma unroll
    for (int nt = 0; nt < 4; ++nt)
      bf[nt] = *(const v16h*)(Wbase + (size_t)(nt * 16 + n0) * D_ + k0 + kbB);
    #pragma unroll
    for (int ms = 0; ms < 4; ++ms)
      #pragma unroll
      for (int nt = 0; nt < 4; ++nt)
        acc[ms][nt] = wmma16(a[ms].v, bf[nt], acc[ms][nt]);
  }

  if (which == 2) {
    // transposed store: per lane 8 contiguous t-positions -> one 16B store
    _Float16* obase = VpT + (size_t)(b * H_ + h) * HS_ * T_;
    #pragma unroll
    for (int ms = 0; ms < 4; ++ms)
      #pragma unroll
      for (int nt = 0; nt < 4; ++nt) {
        const int hs = nt * 16 + n0;
        v8h pk;
        #pragma unroll
        for (int r = 0; r < 8; ++r) pk[r] = (_Float16)acc[ms][nt][r];
        *(v8h*)(obase + (size_t)hs * T_ + tBase + ms * 16 + hi * 8) = pk;
      }
  } else {
    const float sc = (which == 0) ? 0.125f : 1.0f;  // HS^-0.5 folded into Q
    _Float16* ob0 =
        ((which == 0) ? Qp : Kp) + ((size_t)(b * H_ + h) * T_ + tBase) * HS_;
    #pragma unroll
    for (int ms = 0; ms < 4; ++ms)
      #pragma unroll
      for (int nt = 0; nt < 4; ++nt) {
        const int hs = nt * 16 + n0;
        #pragma unroll
        for (int r = 0; r < 8; ++r)
          ob0[(size_t)(ms * 16 + r + hi * 8) * HS_ + hs] =
              (_Float16)(acc[ms][nt][r] * sc);
      }
  }
}

// ----------------------------------------------------- causal flash attention
// Workgroup = 4 waves = 64 consecutive queries of one (b,h). K/V tiles of 32
// positions are DMA'd into LDS once per workgroup by the Tensor Data Mover
// (double-buffered, TENSORcnt + workgroup barriers), then all 4 waves consume
// them from LDS. Each wave owns 16 queries; skips tiles beyond its causal
// limit via wave-uniform branches (EXEC stays all-ones for WMMA).
__global__ void __launch_bounds__(128) flash_kernel(
    const _Float16* __restrict__ Qp, const _Float16* __restrict__ Kp,
    const _Float16* __restrict__ VpT, _Float16* __restrict__ Oc) {
  __shared__ _Float16 ldsK[2][32 * 64];  // [buf][kv][hs]   4KB each
  __shared__ _Float16 ldsV[2][64 * 32];  // [buf][hs][kv]   4KB each
  __shared__ _Float16 ldsP[4][16 * 32];  // per-wave P tile 1KB each

  const int lane = threadIdx.x & 31;
  const int wave = threadIdx.x >> 5;

  const int gw = blockIdx.x;                  // B_*H_*(T_/64) blocks
  const int b  = gw / (H_ * (T_ / 64));
  const int h  = (gw / (T_ / 64)) % H_;
  const int mt = gw % (T_ / 64);
  const int tBase = mt * 64;                  // block query base
  const int tW    = tBase + wave * 16;        // this wave's query base

  const int n0  = lane & 15;
  const int hi  = lane >> 4;
  const int kbA = hi ? 8  : 0;
  const int kbB = hi ? 16 : 0;

  const size_t bh = (size_t)b * H_ + h;
  const _Float16* Qrow  = Qp + (bh * T_ + tW + n0) * HS_;
  const _Float16* Kbase = Kp + bh * T_ * HS_;
  const _Float16* Vbase = VpT + bh * HS_ * T_;

  ABFrag aq[2];
  #pragma unroll
  for (int c = 0; c < 2; ++c) {
    aq[c].h[0] = *(const v8h*)(Qrow + 32 * c + kbB);
    aq[c].h[1] = *(const v8h*)(Qrow + 32 * c + kbB + 8);
  }

  float mrun[8], lrun[8];
  v8f acc[4];
  #pragma unroll
  for (int r = 0; r < 8; ++r) { mrun[r] = -__builtin_inff(); lrun[r] = 0.f; }
  #pragma unroll
  for (int nt = 0; nt < 4; ++nt)
    #pragma unroll
    for (int r = 0; r < 8; ++r) acc[nt][r] = 0.f;

  _Float16* prow = &ldsP[wave][0];

  const int nkvBlk = tBase / 32 + 2;          // block-wide causal tile count
  const int nkvW   = (tW + 15) / 32 + 1;      // this wave's causal tile count

  // Preload tile 0 into buffer 0 (wave 0 drives the TDM).
  if (wave == 0) {
    tdm_load_2d_f16((unsigned int)(unsigned long long)&ldsK[0][0],
                    Kbase, HS_, T_, HS_, 32, HS_);
    tdm_load_2d_f16((unsigned int)(unsigned long long)&ldsV[0][0],
                    Vbase, T_, HS_, 32, HS_, T_);
  }

  for (int kt = 0; kt < nkvBlk; ++kt) {
    const int kv0 = kt * 32;
    const int buf = kt & 1;

    __syncthreads();  // everyone done reading the buffer we're about to fill
    if (wave == 0) {
      if (kt + 1 < nkvBlk) {
        const int kv1 = (kt + 1) * 32;
        const int nb = (kt + 1) & 1;
        tdm_load_2d_f16((unsigned int)(unsigned long long)&ldsK[nb][0],
                        Kbase + (size_t)kv1 * HS_, HS_, T_, HS_, 32, HS_);
        tdm_load_2d_f16((unsigned int)(unsigned long long)&ldsV[nb][0],
                        Vbase + kv1, T_, HS_, 32, HS_, T_);
        __builtin_amdgcn_s_wait_tensorcnt(2);  // current tile's 2 DMAs done
      } else {
        __builtin_amdgcn_s_wait_tensorcnt(0);
      }
    }
    __syncthreads();  // current buffer is valid for all waves

    if (kt < nkvW) {
      const _Float16* kbuf = &ldsK[buf][0];
      const _Float16* vbuf = &ldsV[buf][0];

      v8f s0, s1;
      #pragma unroll
      for (int r = 0; r < 8; ++r) { s0[r] = 0.f; s1[r] = 0.f; }
      #pragma unroll
      for (int c = 0; c < 2; ++c) {
        ABFrag b0, b1;
        const _Float16* k0r = kbuf + (size_t)(n0)*HS_ + 32 * c + kbB;
        const _Float16* k1r = kbuf + (size_t)(16 + n0) * HS_ + 32 * c + kbB;
        b0.h[0] = *(const v8h*)(k0r);
        b0.h[1] = *(const v8h*)(k0r + 8);
        b1.h[0] = *(const v8h*)(k1r);
        b1.h[1] = *(const v8h*)(k1r + 8);
        s0 = wmma16(aq[c].v, b0.v, s0);
        s1 = wmma16(aq[c].v, b1.v, s1);
      }
      // causal mask + row stats (rows live across 16-lane half-waves)
      float f0[8], f1[8], tmax[8];
      #pragma unroll
      for (int r = 0; r < 8; ++r) {
        const int tq = tW + r + hi * 8;
        f0[r] = (kv0 + n0 <= tq)      ? s0[r] : -__builtin_inff();
        f1[r] = (kv0 + 16 + n0 <= tq) ? s1[r] : -__builtin_inff();
        tmax[r] = fmaxf(f0[r], f1[r]);
      }
      #pragma unroll
      for (int off = 1; off < 16; off <<= 1)
        #pragma unroll
        for (int r = 0; r < 8; ++r)
          tmax[r] = fmaxf(tmax[r], __shfl_xor(tmax[r], off, 32));
      float fac[8], psum[8];
      #pragma unroll
      for (int r = 0; r < 8; ++r) {
        const float mnew = fmaxf(mrun[r], tmax[r]);
        fac[r] = __expf(mrun[r] - mnew);
        mrun[r] = mnew;
        f0[r] = __expf(f0[r] - mnew);
        f1[r] = __expf(f1[r] - mnew);
        psum[r] = f0[r] + f1[r];
      }
      #pragma unroll
      for (int off = 1; off < 16; off <<= 1)
        #pragma unroll
        for (int r = 0; r < 8; ++r)
          psum[r] += __shfl_xor(psum[r], off, 32);
      #pragma unroll
      for (int r = 0; r < 8; ++r) lrun[r] = lrun[r] * fac[r] + psum[r];
      #pragma unroll
      for (int nt = 0; nt < 4; ++nt)
        #pragma unroll
        for (int r = 0; r < 8; ++r) acc[nt][r] *= fac[r];

      // Re-layout P: C-layout -> LDS -> A-layout (in-order per-wave LDS)
      #pragma unroll
      for (int r = 0; r < 8; ++r) {
        const int m = r + hi * 8;
        prow[m * 32 + n0]      = (_Float16)f0[r];
        prow[m * 32 + 16 + n0] = (_Float16)f1[r];
      }
      asm volatile("s_wait_dscnt 0" ::: "memory");
      ABFrag pa;
      pa.h[0] = *(const v8h*)(prow + n0 * 32 + kbB);
      pa.h[1] = *(const v8h*)(prow + n0 * 32 + kbB + 8);

      #pragma unroll
      for (int nt = 0; nt < 4; ++nt) {
        ABFrag bv;
        const _Float16* vr = vbuf + (size_t)(nt * 16 + n0) * 32 + kbB;
        bv.h[0] = *(const v8h*)(vr);
        bv.h[1] = *(const v8h*)(vr + 8);
        acc[nt] = wmma16(pa.v, bv.v, acc[nt]);
      }
    }
  }

  float inv[8];
  #pragma unroll
  for (int r = 0; r < 8; ++r) inv[r] = 1.0f / lrun[r];
  _Float16* obase = Oc + ((size_t)b * T_ + tW) * D_ + h * HS_;
  #pragma unroll
  for (int nt = 0; nt < 4; ++nt) {
    const int col = nt * 16 + n0;
    #pragma unroll
    for (int r = 0; r < 8; ++r)
      obase[(size_t)(r + hi * 8) * D_ + col] = (_Float16)(acc[nt][r] * inv[r]);
  }
}

// ------------------------------------------------ output projection + bias
// 64x64 wave tile, same register blocking as proj_kernel.
__global__ void __launch_bounds__(256) outproj_kernel(
    const _Float16* __restrict__ Oc, const _Float16* __restrict__ WoT,
    const float* __restrict__ bo, float* __restrict__ out) {
  const int lane = threadIdx.x & 31;
  const int wave = threadIdx.x >> 5;
  const int gw = blockIdx.x * 8 + wave;
  const int mt = gw / (D_ / 64);
  const int ng = gw % (D_ / 64);
  const int rowBase = mt * 64;  // flat over B_*T_

  const int n0  = lane & 15;
  const int hi  = lane >> 4;
  const int kbA = hi ? 8  : 0;
  const int kbB = hi ? 16 : 0;

  v8f acc[4][4];
  #pragma unroll
  for (int ms = 0; ms < 4; ++ms)
    #pragma unroll
    for (int nt = 0; nt < 4; ++nt)
      #pragma unroll
      for (int r = 0; r < 8; ++r) acc[ms][nt][r] = 0.f;

  for (int kc = 0; kc < D_ / 32; ++kc) {
    const int k0 = kc * 32;
    ABFrag a[4];
    #pragma unroll
    for (int ms = 0; ms < 4; ++ms) {
      const _Float16* Arow = Oc + (size_t)(rowBase + ms * 16 + n0) * D_;
      a[ms].h[0] = *(const v8h*)(Arow + k0 + kbB);
      a[ms].h[1] = *(const v8h*)(Arow + k0 + kbB + 8);
    }
    v16h bf[4];
    #pragma unroll
    for (int nt = 0; nt < 4; ++nt)
      bf[nt] =
          *(const v16h*)(WoT + (size_t)(ng * 64 + nt * 16 + n0) * D_ + k0 + kbB);
    #pragma unroll
    for (int ms = 0; ms < 4; ++ms)
      #pragma unroll
      for (int nt = 0; nt < 4; ++nt)
        acc[ms][nt] = wmma16(a[ms].v, bf[nt], acc[ms][nt]);
  }

  #pragma unroll
  for (int ms = 0; ms < 4; ++ms)
    #pragma unroll
    for (int nt = 0; nt < 4; ++nt) {
      const int col = ng * 64 + nt * 16 + n0;
      const float bias = bo[col];
      #pragma unroll
      for (int r = 0; r < 8; ++r)
        out[(size_t)(rowBase + ms * 16 + r + hi * 8) * D_ + col] =
            acc[ms][nt][r] + bias;
    }
}

// ---------------------------------------------------------------------------
extern "C" void kernel_launch(void* const* d_in, const int* in_sizes, int n_in,
                              void* d_out, int out_size, void* d_ws,
                              size_t ws_size, hipStream_t stream) {
  (void)in_sizes; (void)n_in; (void)out_size; (void)ws_size;
  const float* q  = (const float*)d_in[0];
  const float* k  = (const float*)d_in[1];
  const float* v  = (const float*)d_in[2];
  const float* Wq = (const float*)d_in[3];
  const float* Wk = (const float*)d_in[4];
  const float* Wv = (const float*)d_in[5];
  const float* Wo = (const float*)d_in[6];
  const float* bo = (const float*)d_in[7];
  // d_in[8] = causal mask: recomputed analytically in-kernel.
  float* out = (float*)d_out;

  const size_t nQKV = (size_t)B_ * T_ * D_;       // 4,194,304
  const size_t nW   = (size_t)H_ * D_ * HS_;      // 1,048,576 (== D*D)
  const size_t nP   = (size_t)B_ * H_ * T_ * HS_; // 4,194,304

  _Float16* p   = (_Float16*)d_ws;  // total 64 MiB of f16 scratch
  _Float16* qh  = p; p += nQKV;
  _Float16* kh  = p; p += nQKV;
  _Float16* vh  = p; p += nQKV;
  _Float16* WqT = p; p += nW;
  _Float16* WkT = p; p += nW;
  _Float16* WvT = p; p += nW;
  _Float16* WoT = p; p += nW;
  _Float16* Qp  = p; p += nP;
  _Float16* Kp  = p; p += nP;
  _Float16* VpT = p; p += nP;
  _Float16* Oc  = p; p += nQKV;

  cast_qkv_kernel<<<(int)((nQKV + 255) / 256), 256, 0, stream>>>(q, k, v, qh,
                                                                 kh, vh);
  cast_w_kernel<<<(int)((nW + 255) / 256), 256, 0, stream>>>(
      Wq, Wk, Wv, Wo, WqT, WkT, WvT, WoT);
  proj_kernel<<<(3 * B_ * H_ * (T_ / 64)) / 8, 256, 0, stream>>>(
      qh, kh, vh, WqT, WkT, WvT, Qp, Kp, VpT);
  flash_kernel<<<B_ * H_ * (T_ / 64), 128, 0, stream>>>(Qp, Kp, VpT, Oc);
  outproj_kernel<<<((B_ * T_ / 64) * (D_ / 64)) / 8, 256, 0, stream>>>(
      Oc, WoT, bo, out);
}
